// MPNet_2396591751358
// MI455X (gfx1250) — matrix operand order repaired and checked by reading, not compiled, not run on hardware
//
#include <hip/hip_runtime.h>
#include <hip/hip_bf16.h>

typedef __attribute__((ext_vector_type(16))) _Float16 v16h;
typedef __attribute__((ext_vector_type(8)))  float    v8f;

#define NN 100000
#define NE 1000000
#define D  64

// ---------------------------------------------------------------- zero
__global__ void zero_kernel(float* __restrict__ p, int n4) {
  float4 z = {0.f, 0.f, 0.f, 0.f};
  for (int i = blockIdx.x * blockDim.x + threadIdx.x; i < n4;
       i += gridDim.x * blockDim.x)
    ((float4*)p)[i] = z;
}

// ---------------------------------------------------------------- weight pre-pack
// Pack B = [Wrel ; root] (128x64 f32) into the WMMA B-operand layout as f16:
// for tile (kc, jt), lane l holds column j = 16*jt + (l%16), rows
// kbase = 32*(kc&1) + 16*(l/16) .. +15, packed as sequential f16 pairs.
// Output DWORD index: ((kc*4 + jt)*32 + lane)*8 + v  (v = packed pair index).
// 4096 threads per layer; runs once per launch, amortized over 6250 GEMM waves.
__global__ void pack_weights(const float* __restrict__ Wrel,
                             const float* __restrict__ root,
                             unsigned int* __restrict__ bp) {
  const int tid = blockIdx.x * blockDim.x + threadIdx.x;
  if (tid >= 4096) return;
  const int v    = tid & 7;
  const int lane = (tid >> 3) & 31;
  const int jt   = (tid >> 8) & 3;
  const int kc   = (tid >> 10) & 3;
  const int r = lane & 15, h = lane >> 4;
  const float* srcB = (kc < 2) ? Wrel : root;
  const int krow = 32 * (kc & 1) + 16 * h;
  const int j = 16 * jt + r;
  union { _Float16 f[2]; unsigned int u; } pk;
  pk.f[0] = (_Float16)srcB[(krow + 2 * v)     * D + j];
  pk.f[1] = (_Float16)srcB[(krow + 2 * v + 1) * D + j];
  bp[tid] = pk.u;
}

// ---------------------------------------------------------------- scatter
// agg[src] += x[dst] for edges with etype == rel. One wave per edge:
// relation test is wave-uniform, 64-float row = 2 floats/lane (coalesced),
// atomics resolve in L2 (agg is 25.6MB, L2-resident).
__global__ void scatter_kernel(const float* __restrict__ xin,
                               const int* __restrict__ src,
                               const int* __restrict__ dst,
                               const int* __restrict__ etype,
                               int rel, int nEdges,
                               float* __restrict__ agg) {
  const int lane   = threadIdx.x & 31;
  const int wave   = (blockIdx.x * blockDim.x + threadIdx.x) >> 5;
  const int nWaves = (gridDim.x * blockDim.x) >> 5;
  for (int e = wave; e < nEdges; e += nWaves) {
    int ef = e + 4 * nWaves;                // prefetch edge metadata ahead
    if (ef < nEdges) {
      __builtin_prefetch(&etype[ef], 0, 0);
      __builtin_prefetch(&dst[ef], 0, 0);
      __builtin_prefetch(&src[ef], 0, 0);
    }
    if (etype[e] == rel) {
      const int s = src[e], d = dst[e];
      const float* row  = xin + (long)d * D;
      float*       arow = agg + (long)s * D;
      float v0 = row[lane];
      float v1 = row[lane + 32];
      atomicAdd(arow + lane,      v0);
      atomicAdd(arow + lane + 32, v1);
    }
  }
}

// ---------------------------------------------------------------- update
// xout = relu( [agg | xin] @ [Wrel ; root] + bias ),  K = 128, N = 64.
// One wave per 16-node tile; 4 K-chunks x 4 N-tiles of v_wmma_f32_16x16x32_f16.
// B comes pre-packed in lane layout: one 32B contiguous load per lane per tile.
__global__ void __launch_bounds__(128)
update_kernel(const float* __restrict__ agg, const float* __restrict__ xin,
              const v16h* __restrict__ bpack,       // [16 tiles][32 lanes] v16h
              const float* __restrict__ bias, float* __restrict__ xout,
              int nNodes) {
  const int lane     = threadIdx.x & 31;
  const int waveId   = threadIdx.x >> 5;
  const int r        = lane & 15;       // row (A) / column (B,C)
  const int h        = lane >> 4;       // lane half
  const int tileBase = blockIdx.x * 64 + waveId * 16;

  int nodeA = tileBase + r;
  if (nodeA >= nNodes) nodeA = nNodes - 1;   // clamp: EXEC must stay all-1s

  v8f c[4];
  #pragma unroll
  for (int jt = 0; jt < 4; ++jt) c[jt] = (v8f){0.f,0.f,0.f,0.f,0.f,0.f,0.f,0.f};

  #pragma unroll
  for (int kc = 0; kc < 4; ++kc) {
    // A chunk: K global [32kc, 32kc+32). kc<2 -> agg, kc>=2 -> xin.
    // lane holds row r, local K = {16h..16h+15}, packed in order.
    const float* srcA = (kc < 2) ? (agg + (long)nodeA * D + 32 * kc)
                                 : (xin + (long)nodeA * D + 32 * (kc - 2));
    v16h a;
    #pragma unroll
    for (int i = 0; i < 8; ++i) a[i]     = (_Float16)srcA[16 * h + i];
    #pragma unroll
    for (int i = 0; i < 8; ++i) a[8 + i] = (_Float16)srcA[16 * h + 8 + i];

    #pragma unroll
    for (int jt = 0; jt < 4; ++jt) {
      v16h b = bpack[(kc * 4 + jt) * 32 + lane];   // 32B coalesced, pre-swizzled
      c[jt] = __builtin_amdgcn_wmma_f32_16x16x32_f16(
          /*neg_a=*/false, a, /*neg_b=*/false, b,
          /*c_mod=*/(short)0, c[jt], /*reuse_a=*/false, /*reuse_b=*/false);
    }
  }

  // C/D layout: VGPR i, lane -> (M = i + 8h, N = r). Bias + ReLU + store.
  #pragma unroll
  for (int jt = 0; jt < 4; ++jt) {
    const int j = 16 * jt + r;
    const float bj = bias[j];
    #pragma unroll
    for (int i = 0; i < 8; ++i) {
      const int node = tileBase + i + 8 * h;
      if (node < nNodes) {
        float v = c[jt][i] + bj;
        xout[(long)node * D + j] = v > 0.f ? v : 0.f;
      }
    }
  }
}

// ---------------------------------------------------------------- final
// logits = x @ Wl + bl ; out = log_softmax(logits). 0.2 GFLOP total -> scalar.
__global__ void final_kernel(const float* __restrict__ x,
                             const float* __restrict__ Wl,
                             const float* __restrict__ bl,
                             float* __restrict__ out, int n) {
  const int node = blockIdx.x * blockDim.x + threadIdx.x;
  if (node >= n) return;
  float xr[64];
  const float4* xp = (const float4*)(x + (long)node * D);
  #pragma unroll
  for (int i = 0; i < 16; ++i) {
    float4 t = xp[i];
    xr[4*i] = t.x; xr[4*i+1] = t.y; xr[4*i+2] = t.z; xr[4*i+3] = t.w;
  }
  float lg[16];
  #pragma unroll
  for (int j = 0; j < 16; ++j) {
    float acc = bl[j];
    #pragma unroll
    for (int k = 0; k < 64; ++k) acc = fmaf(xr[k], Wl[k * 16 + j], acc);
    lg[j] = acc;
  }
  float m = lg[0];
  #pragma unroll
  for (int j = 1; j < 16; ++j) m = fmaxf(m, lg[j]);
  float s = 0.f;
  #pragma unroll
  for (int j = 0; j < 16; ++j) s += __expf(lg[j] - m);
  const float lse = m + __logf(s);
  #pragma unroll
  for (int j = 0; j < 16; ++j) out[(long)node * 16 + j] = lg[j] - lse;
}

// ---------------------------------------------------------------- launch
extern "C" void kernel_launch(void* const* d_in, const int* in_sizes, int n_in,
                              void* d_out, int out_size, void* d_ws, size_t ws_size,
                              hipStream_t stream) {
  const float* x     = (const float*)d_in[0];
  const int*   eidx  = (const int*)d_in[1];   // [2, NE]
  const int*   etype = (const int*)d_in[2];   // [NE]
  const float* W1    = (const float*)d_in[3]; // [5,64,64]
  const float* root1 = (const float*)d_in[4];
  const float* b1    = (const float*)d_in[5];
  const float* W2    = (const float*)d_in[6]; // [5,64,64]
  const float* root2 = (const float*)d_in[7];
  const float* b2    = (const float*)d_in[8];
  const float* Wl    = (const float*)d_in[9]; // [64,16]
  const float* bl    = (const float*)d_in[10];
  float* out = (float*)d_out;

  float* agg = (float*)d_ws;                  // [NN*D] f32
  float* h1  = agg + (size_t)NN * D;          // [NN*D] f32
  float* h2  = h1  + (size_t)NN * D;          // [NN*D] f32
  unsigned int* bp0 = (unsigned int*)(h2 + (size_t)NN * D); // 4096 DW per layer
  unsigned int* bp1 = bp0 + 4096;
  unsigned int* bp2 = bp1 + 4096;

  const int* src = eidx;           // messages aggregated at edge_index[0]
  const int* dst = eidx + NE;      // gathered from edge_index[1]

  const int n4    = NN * D / 4;
  const int zGrid = 1024, zBlk = 256;
  const int sGrid = 2048, sBlk = 256;
  const int uGrid = (NN + 63) / 64, uBlk = 128;

  // Pre-swizzle all three layers' weights into WMMA B layout (f16).
  pack_weights<<<16, 256, 0, stream>>>(W1 + 0 * D * D, root1, bp0);
  pack_weights<<<16, 256, 0, stream>>>(W2 + 1 * D * D, root2, bp1);
  pack_weights<<<16, 256, 0, stream>>>(W2 + 2 * D * D, root2, bp2);

  // Layer 0: rel=0, in = x, out = h1
  zero_kernel   <<<zGrid, zBlk, 0, stream>>>(agg, n4);
  scatter_kernel<<<sGrid, sBlk, 0, stream>>>(x, src, dst, etype, 0, NE, agg);
  update_kernel <<<uGrid, uBlk, 0, stream>>>(agg, x, (const v16h*)bp0, b1, h1, NN);

  // Layer 1: rel=1, in = h1, out = h2
  zero_kernel   <<<zGrid, zBlk, 0, stream>>>(agg, n4);
  scatter_kernel<<<sGrid, sBlk, 0, stream>>>(h1, src, dst, etype, 1, NE, agg);
  update_kernel <<<uGrid, uBlk, 0, stream>>>(agg, h1, (const v16h*)bp1, b2, h2, NN);

  // Layer 2: rel=2, in = h2, out = h1 (buffer reuse)
  zero_kernel   <<<zGrid, zBlk, 0, stream>>>(agg, n4);
  scatter_kernel<<<sGrid, sBlk, 0, stream>>>(h2, src, dst, etype, 2, NE, agg);
  update_kernel <<<uGrid, uBlk, 0, stream>>>(agg, h2, (const v16h*)bp2, b2, h1, NN);

  // Final linear + log_softmax
  final_kernel<<<(NN + 255) / 256, 256, 0, stream>>>(h1, Wl, bl, out, NN);
}
